// G_GNN_61435212202442
// MI455X (gfx1250) — matrix operand order repaired and checked by reading, not compiled, not run on hardware
//
#include <hip/hip_runtime.h>
#include <cstring>

typedef __attribute__((ext_vector_type(16))) _Float16 v16h;
typedef __attribute__((ext_vector_type(8)))  _Float16 v8h;
typedef __attribute__((ext_vector_type(8)))  float    v8f;

// ---------------- helpers ----------------
__device__ __forceinline__ int f2o_d(float f) {           // order-preserving float->int
  int i = __float_as_int(f);
  return i >= 0 ? i : (i ^ 0x7fffffff);
}
__device__ __forceinline__ float o2f_d(int i) {
  return __int_as_float(i >= 0 ? i : (i ^ 0x7fffffff));
}
__device__ __forceinline__ float wred(float v) {
#pragma unroll
  for (int off = 16; off > 0; off >>= 1) v += __shfl_xor(v, off, 32);
  return v;
}

// ---------------- fill / convert kernels ----------------
__global__ __launch_bounds__(256) void k_fill_f32(float* __restrict__ p, float v, long n) {
  long i = (long)blockIdx.x * blockDim.x + threadIdx.x;
  if (i < n) p[i] = v;
}
__global__ __launch_bounds__(256) void k_fill_i32(int* __restrict__ p, int v, long n) {
  long i = (long)blockIdx.x * blockDim.x + threadIdx.x;
  if (i < n) p[i] = v;
}
__global__ __launch_bounds__(256) void k_cvt_f2h(_Float16* __restrict__ o,
                                                 const float* __restrict__ p, long n) {
  long i = (long)blockIdx.x * blockDim.x + threadIdx.x;
  if (i < n) o[i] = (_Float16)p[i];
}

// Convert row-major f32 weight W[K][Nw] (placed at column offset colOff of the
// combined B) into the WMMA 16x16x32 B-fragment layout, f16:
// frag[(((ct*nK)+ks)*32+lane)*16+hj] = W[ks*32 + (lane>>4)*8 + (hj&7) + (hj>>3)*16][ct*16 + (lane&15)]
__global__ __launch_bounds__(256) void k_prep_frag(
    _Float16* __restrict__ frag, const float* __restrict__ W,
    int K, int Nw, int colOff, int Kpad) {
  int nK = Kpad >> 5;
  int nCTl = (Nw + 15) >> 4;
  long total = (long)nCTl * nK * 512;
  long t = (long)blockIdx.x * blockDim.x + threadIdx.x;
  if (t >= total) return;
  int hj = (int)(t & 15);
  int lane = (int)((t >> 4) & 31);
  long rest = t >> 9;
  int ks = (int)(rest % nK);
  int ctl = (int)(rest / nK);
  int kb = (lane >> 4) * 8;
  int col = lane & 15;
  int k = ks * 32 + kb + (hj & 7) + ((hj >> 3) << 4);
  int lcol = ctl * 16 + col;
  float v = 0.f;
  if (k < K && lcol < Nw) v = W[(long)k * Nw + lcol];
  int ct = (colOff >> 4) + ctl;
  frag[((((long)ct * nK) + ks) * 32 + lane) * 16 + hj] = (_Float16)v;
}

// ---------------- gathered GEMM via WMMA ----------------
// out[R, NoutStore] = concat-gather(f16 A segments)[R, Kpad] @ Bfrag (+bias, relu)
// Segments {base(f16), idx, width, rowstride}; idx==null -> row id. Widths are
// multiples of 8 so staging moves aligned 16-byte chunks.
__global__ __launch_bounds__(128) void k_gemm_gather(
    int R, int Kpad, int Ncols,
    const _Float16* __restrict__ Bfrag,
    void* __restrict__ outv, int outStride, int NoutStore, int outHalf,
    const float* __restrict__ bias, int relu,
    const _Float16* __restrict__ s0, const int* __restrict__ i0, int w0, int ld0,
    const _Float16* __restrict__ s1, const int* __restrict__ i1, int w1, int ld1,
    const _Float16* __restrict__ s2, const int* __restrict__ i2, int w2, int ld2,
    const _Float16* __restrict__ s3, const int* __restrict__ i3, int w3, int ld3) {
  extern __shared__ _Float16 As[];      // [16][Kpad] f16 staged A tile
  __shared__ int rowsrc[4][16];
  const int tid = threadIdx.x;
  const int r0 = blockIdx.x * 16;

  if (tid < 64) {
    int seg = tid >> 4, rr = tid & 15;
    int gr = r0 + rr;
    int gi = (gr < R) ? gr : 0;
    const int* ip = (seg == 0) ? i0 : (seg == 1) ? i1 : (seg == 2) ? i2 : i3;
    rowsrc[seg][rr] = ip ? ip[gi] : gi;
  }
  __syncthreads();

  const int nCh = Kpad >> 3;            // 8-half chunks per row
  const int tot = 16 * nCh;
  for (int f = tid; f < tot; f += 128) {
    int rr = f / nCh;
    int ck = f - rr * nCh;
    int k0 = ck << 3;
    const _Float16* sp = nullptr;
    long sidx = 0;
    if (r0 + rr < R) {
      int k = k0;
      if (k < w0)              { sp = s0; sidx = (long)rowsrc[0][rr] * ld0 + k; }
      else if ((k -= w0) < w1) { sp = s1; sidx = (long)rowsrc[1][rr] * ld1 + k; }
      else if ((k -= w1) < w2) { sp = s2; sidx = (long)rowsrc[2][rr] * ld2 + k; }
      else if ((k -= w2) < w3) { sp = s3; sidx = (long)rowsrc[3][rr] * ld3 + k; }
    }
    v8h val;
    if (sp) {
      val = *(const v8h*)(sp + sidx);
    } else {
#pragma unroll
      for (int j = 0; j < 8; ++j) val[j] = (_Float16)0.f;
    }
    *(v8h*)(As + rr * Kpad + k0) = val;
  }
  __syncthreads();

  const int lane = tid & 31;
  const int wv = tid >> 5;
  const int nK = Kpad >> 5;
  const int nCT = Ncols >> 4;
  const int row16 = lane & 15;
  const int hiHalf = lane >> 4;
  const int koff = hiHalf * 8;
  const v16h* __restrict__ Bf = (const v16h*)Bfrag;

  for (int ct = wv; ct < nCT; ct += 4) {
    const int gcol = ct * 16 + row16;
    float bv = 0.f;
    if (bias != nullptr && gcol < NoutStore) bv = bias[gcol];
    v8f acc;
#pragma unroll
    for (int q = 0; q < 8; ++q) acc[q] = bv;

    for (int ks = 0; ks < nK; ++ks) {
      // A fragment: lane<16 -> K[0..7] & K[16..23]; lane>=16 -> K[8..15] & K[24..31]
      const _Float16* abase = As + row16 * Kpad + ks * 32;
      v8h alo = *(const v8h*)(abase + koff);
      v8h ahi = *(const v8h*)(abase + 16 + koff);
      v16h a;
#pragma unroll
      for (int j = 0; j < 8; ++j) { a[j] = alo[j]; a[j + 8] = ahi[j]; }
      v16h b = Bf[((long)ct * nK + ks) * 32 + lane];
      acc = __builtin_amdgcn_wmma_f32_16x16x32_f16(false, a, false, b,
                                                   (short)0, acc, false, false);
    }
#pragma unroll
    for (int q = 0; q < 8; ++q) {
      int row = r0 + q + 8 * hiHalf;          // C/D layout: M = q + 8*(lane>=16)
      if (row < R && gcol < NoutStore) {
        float o = acc[q];
        if (relu && o < 0.f) o = 0.f;
        if (outHalf) ((_Float16*)outv)[(long)row * outStride + gcol] = (_Float16)o;
        else         ((float*)outv)[(long)row * outStride + gcol] = o;
      }
    }
  }
}

// ---------------- attention kernels (f16 features, f32 math) ----------------
__global__ __launch_bounds__(128) void k_att_logits(
    int E_, int H, int nh, int HC,
    const _Float16* __restrict__ hN,
    const _Float16* __restrict__ ep, int ldEp,
    const int* __restrict__ src, const int* __restrict__ dst,
    const float* __restrict__ av_s, const float* __restrict__ av_d,
    const float* __restrict__ av_e,
    float* __restrict__ lg, int* __restrict__ mxbits) {
  int e = blockIdx.x * 4 + (threadIdx.x >> 5);
  if (e >= E_) return;
  int lane = threadIdx.x & 31;
  int sN = src[e], dN = dst[e];
  const _Float16* hs = hN + (long)sN * HC;
  const _Float16* hdp = hN + (long)dN * HC;
  const _Float16* er = ep + (long)e * ldEp;
  for (int h = 0; h < H; ++h) {
    float p = 0.f;
    for (int c = lane; c < nh; c += 32) {
      int cc = h * nh + c;
      p += (float)hs[cc] * av_s[cc] + (float)hdp[cc] * av_d[cc] + (float)er[cc] * av_e[cc];
    }
    p = wred(p);
    if (lane == 0) {
      float v = (p > 0.f) ? p : 0.2f * p;      // leaky relu 0.2
      lg[(long)e * H + h] = v;
      atomicMax(&mxbits[dN * H + h], f2o_d(v));
    }
  }
}

__global__ __launch_bounds__(256) void k_att_expsum(
    long EH, int H, const int* __restrict__ dst,
    const int* __restrict__ mxbits, float* __restrict__ lg,
    float* __restrict__ ssum) {
  long t = (long)blockIdx.x * blockDim.x + threadIdx.x;
  if (t >= EH) return;
  int e = (int)(t / H), h = (int)(t % H);
  int dN = dst[e];
  float m = o2f_d(mxbits[dN * H + h]);
  float v = __expf(lg[t] - m);
  lg[t] = v;
  atomicAdd(&ssum[dN * H + h], v);
}

__global__ __launch_bounds__(128) void k_att_scatter(
    int E_, int H, int nh, int HC,
    const _Float16* __restrict__ hN,
    const _Float16* __restrict__ ep, int ldEp,
    const int* __restrict__ src, const int* __restrict__ dst,
    const float* __restrict__ lg, const float* __restrict__ ssum,
    float* __restrict__ z) {
  int e = blockIdx.x * 4 + (threadIdx.x >> 5);
  if (e >= E_) return;
  int lane = threadIdx.x & 31;
  int sN = src[e], dN = dst[e];
  for (int c = lane; c < HC; c += 32) {
    int h = c / nh;
    float alpha = lg[(long)e * H + h] / (ssum[dN * H + h] + 1e-16f);
    float m = alpha * ((float)hN[(long)sN * HC + c] + (float)ep[(long)e * ldEp + c]);
    atomicAdd(&z[(long)dN * HC + c], m);
  }
}

// ---------------- host ----------------
extern "C" void kernel_launch(void* const* d_in, const int* in_sizes, int n_in,
                              void* d_out, int out_size, void* d_ws, size_t ws_size,
                              hipStream_t stream) {
  (void)n_in; (void)out_size; (void)ws_size;
  const float* x   = (const float*)d_in[0];
  const float* eat = (const float*)d_in[1];
  const int* src   = (const int*)d_in[2];
  const int* dst   = (const int*)d_in[3];
  const int* lsrc  = (const int*)d_in[4];
  const int* ldst  = (const int*)d_in[5];
  const int* le1   = (const int*)d_in[6];
  const int* le2   = (const int*)d_in[7];
  const float* W1n  = (const float*)d_in[8];
  const float* W1ea = (const float*)d_in[9];
  const float* W1eo = (const float*)d_in[10];
  const float* a1s  = (const float*)d_in[11];
  const float* a1d  = (const float*)d_in[12];
  const float* a1e  = (const float*)d_in[13];
  const float* W2n  = (const float*)d_in[14];
  const float* W2ea = (const float*)d_in[15];
  const float* W2eo = (const float*)d_in[16];
  const float* a2s  = (const float*)d_in[17];
  const float* a2d  = (const float*)d_in[18];
  const float* a2e  = (const float*)d_in[19];
  const float* Wd1  = (const float*)d_in[20];
  const float* bd1  = (const float*)d_in[21];
  const float* Wd2  = (const float*)d_in[22];
  const float* bd2  = (const float*)d_in[23];
  float* out = (float*)d_out;

  const int N  = in_sizes[0] / 64;   // 20000
  const int E  = in_sizes[2];        // 320000
  const int EL = in_sizes[4];        // 50000

  char* wsp = (char*)d_ws;
  size_t off = 0;
  auto alloc = [&](size_t bytes) -> void* {
    void* p = wsp + off;
    off = (off + bytes + 255) & ~(size_t)255;
    return p;
  };
  auto fragBytes = [](int Kpad, int Ncols) -> size_t {
    return (size_t)(Ncols / 16) * (size_t)(Kpad / 32) * 512 * 2;
  };

  _Float16* frag1n = (_Float16*)alloc(fragBytes(64, 128));
  _Float16* frag1c = (_Float16*)alloc(fragBytes(160, 192));
  _Float16* frag2n = (_Float16*)alloc(fragBytes(128, 64));
  _Float16* frag2c = (_Float16*)alloc(fragBytes(320, 128));
  _Float16* fragd1 = (_Float16*)alloc(fragBytes(256, 256));
  _Float16* fragd2 = (_Float16*)alloc(fragBytes(256, 16));
  _Float16* xh     = (_Float16*)alloc((size_t)N * 64 * 2);
  _Float16* eath   = (_Float16*)alloc((size_t)E * 16 * 2);
  _Float16* h1h    = (_Float16*)alloc((size_t)N * 128 * 2);
  _Float16* ep1e1h = (_Float16*)alloc((size_t)E * 192 * 2);  // 0..127 ep1 | 128..191 e1
  float*    z1     = (float*)alloc((size_t)N * 128 * 4);
  _Float16* z1h    = (_Float16*)alloc((size_t)N * 128 * 2);
  float*    lg1    = (float*)alloc((size_t)E * 2 * 4);
  int*      mx1    = (int*)alloc((size_t)N * 2 * 4);
  float*    s1     = (float*)alloc((size_t)N * 2 * 4);
  _Float16* h2h    = (_Float16*)alloc((size_t)N * 64 * 2);
  _Float16* ep2e2h = (_Float16*)alloc((size_t)E * 128 * 2);  // 0..63 ep2 | 64..127 e2
  float*    z2     = (float*)alloc((size_t)N * 64 * 4);
  _Float16* z2h    = (_Float16*)alloc((size_t)N * 64 * 2);
  float*    lg2    = (float*)alloc((size_t)E * 4);
  int*      mx2    = (int*)alloc((size_t)N * 4);
  float*    s2     = (float*)alloc((size_t)N * 4);
  _Float16* hdech  = (_Float16*)alloc((size_t)EL * 256 * 2);

  auto prep = [&](_Float16* frag, const float* W, int K, int Nw, int colOff, int Kpad) {
    long total = (long)((Nw + 15) / 16) * (Kpad / 32) * 512;
    k_prep_frag<<<(int)((total + 255) / 256), 256, 0, stream>>>(frag, W, K, Nw, colOff, Kpad);
  };
  prep(frag1n, W1n, 64, 128, 0, 64);
  prep(frag1c, W1ea, 144, 128, 0, 160);
  prep(frag1c, W1eo, 144, 64, 128, 160);
  prep(frag2n, W2n, 128, 64, 0, 128);
  prep(frag2c, W2ea, 320, 64, 0, 320);
  prep(frag2c, W2eo, 320, 64, 64, 320);
  prep(fragd1, Wd1, 256, 256, 0, 256);
  prep(fragd2, Wd2, 256, 3, 0, 256);

  auto cvt = [&](_Float16* o, const float* p, long n) {
    k_cvt_f2h<<<(int)((n + 255) / 256), 256, 0, stream>>>(o, p, n);
  };
  auto gemm = [&](int R, int Kpad, int Ncols, const _Float16* Bf,
                  void* o, int ostr, int nstore, int ohalf, const float* bias, int relu,
                  const _Float16* S0, const int* I0, int w0, int l0,
                  const _Float16* S1, const int* I1, int w1, int l1,
                  const _Float16* S2, const int* I2, int w2, int l2,
                  const _Float16* S3, const int* I3, int w3, int l3) {
    size_t sh = (size_t)16 * Kpad * 2;
    k_gemm_gather<<<(R + 15) / 16, 128, sh, stream>>>(
        R, Kpad, Ncols, Bf, o, ostr, nstore, ohalf, bias, relu,
        S0, I0, w0, l0, S1, I1, w1, l1, S2, I2, w2, l2, S3, I3, w3, l3);
  };

  const _Float16* HN0 = nullptr; const int* IN0 = nullptr;

  float ninf = -3.0e38f;
  int nbits; memcpy(&nbits, &ninf, 4);
  int mxinit = (nbits >= 0) ? nbits : (nbits ^ 0x7fffffff);

  // ---- input conversion ----
  cvt(xh, x, (long)N * 64);
  cvt(eath, eat, (long)E * 16);

  // ---- layer 1 ----
  gemm(N, 64, 128, frag1n, h1h, 128, 128, 1, nullptr, 0,
       xh, nullptr, 64, 64, HN0, IN0, 0, 8, HN0, IN0, 0, 8, HN0, IN0, 0, 8);
  gemm(E, 160, 192, frag1c, ep1e1h, 192, 192, 1, nullptr, 0,
       xh, src, 64, 64, xh, dst, 64, 64, eath, nullptr, 16, 16, HN0, IN0, 0, 8);

  k_fill_i32<<<(int)(((long)N * 2 + 255) / 256), 256, 0, stream>>>(mx1, mxinit, (long)N * 2);
  k_fill_f32<<<(int)(((long)N * 2 + 255) / 256), 256, 0, stream>>>(s1, 0.f, (long)N * 2);
  k_fill_f32<<<(int)(((long)N * 128 + 255) / 256), 256, 0, stream>>>(z1, 0.f, (long)N * 128);

  k_att_logits<<<(E + 3) / 4, 128, 0, stream>>>(E, 2, 64, 128, h1h, ep1e1h, 192, src, dst,
                                                a1s, a1d, a1e, lg1, mx1);
  k_att_expsum<<<(int)(((long)E * 2 + 255) / 256), 256, 0, stream>>>((long)E * 2, 2, dst,
                                                                     mx1, lg1, s1);
  k_att_scatter<<<(E + 3) / 4, 128, 0, stream>>>(E, 2, 64, 128, h1h, ep1e1h, 192, src, dst,
                                                 lg1, s1, z1);
  cvt(z1h, z1, (long)N * 128);

  // ---- layer 2 ----
  gemm(N, 128, 64, frag2n, h2h, 64, 64, 1, nullptr, 0,
       z1h, nullptr, 128, 128, HN0, IN0, 0, 8, HN0, IN0, 0, 8, HN0, IN0, 0, 8);
  gemm(E, 320, 128, frag2c, ep2e2h, 128, 128, 1, nullptr, 0,
       z1h, src, 128, 128, z1h, dst, 128, 128, ep1e1h + 128, nullptr, 64, 192,
       HN0, IN0, 0, 8);

  k_fill_i32<<<(int)(((long)N + 255) / 256), 256, 0, stream>>>(mx2, mxinit, (long)N);
  k_fill_f32<<<(int)(((long)N + 255) / 256), 256, 0, stream>>>(s2, 0.f, (long)N);
  k_fill_f32<<<(int)(((long)N * 64 + 255) / 256), 256, 0, stream>>>(z2, 0.f, (long)N * 64);

  k_att_logits<<<(E + 3) / 4, 128, 0, stream>>>(E, 1, 64, 64, h2h, ep2e2h, 128, src, dst,
                                                a2s, a2d, a2e, lg2, mx2);
  k_att_expsum<<<(int)(((long)E + 255) / 256), 256, 0, stream>>>((long)E, 1, dst,
                                                                 mx2, lg2, s2);
  k_att_scatter<<<(E + 3) / 4, 128, 0, stream>>>(E, 1, 64, 64, h2h, ep2e2h, 128, src, dst,
                                                 lg2, s2, z2);
  cvt(z2h, z2, (long)N * 64);

  // ---- decoder ----
  gemm(EL, 256, 256, fragd1, hdech, 256, 256, 1, bd1, 1,
       z2h, lsrc, 64, 64, z2h, ldst, 64, 64,
       ep2e2h + 64, le1, 64, 128, ep2e2h + 64, le2, 64, 128);
  gemm(EL, 256, 16, fragd2, out, 3, 3, 0, bd2, 0,
       hdech, nullptr, 256, 256, HN0, IN0, 0, 8, HN0, IN0, 0, 8, HN0, IN0, 0, 8);
}
